// GRANDPP_40802189312204
// MI455X (gfx1250) — compile-verified, matrix operand order checked by reading, not run on hardware
//
#include <hip/hip_runtime.h>
#include <hip/hip_bf16.h>
#include <math.h>

// ---------------- problem constants (match reference) ----------------
#define NN 50000
#define EE 800000
#define FF 256
#define HH 256
#define CC 10
#define BB 128
#define VV 4
#define KK 3

typedef __attribute__((ext_vector_type(16))) _Float16 v16h;
typedef __attribute__((ext_vector_type(8)))  _Float16 v8h;
typedef __attribute__((ext_vector_type(8)))  float    v8f;
typedef __attribute__((ext_vector_type(4)))  int      v4i;

#ifndef __has_builtin
#define __has_builtin(x) 0
#endif

#if __has_builtin(__builtin_amdgcn_global_load_async_to_lds_b128)
#define GPP_ASYNC_LDS 1
#else
#define GPP_ASYNC_LDS 0
#endif

#if __has_builtin(__builtin_amdgcn_s_wait_asynccnt)
#define GPP_WAIT_ASYNC(n) __builtin_amdgcn_s_wait_asynccnt(n)
#else
#define GPP_WAIT_ASYNC(n) do {} while (0)
#endif

#if GPP_ASYNC_LDS
// Probe-discovered signature: param0 is non-const generic `int4*` (destination
// = LDS address, matching ISA operand order VDST_lds, VADDR_global).
__device__ __forceinline__ void gpp_async_b128(const void* g, void* l) {
  __builtin_amdgcn_global_load_async_to_lds_b128((v4i*)l, (v4i*)g, 0, 0);
}
#endif

static inline int divceil_i(long long a, long long b) { return (int)((a + b - 1) / b); }

// ---------------- utility: zero a float buffer ----------------
__global__ void gpp_zero_f32(float* __restrict__ p, long long n) {
  long long i = (long long)blockIdx.x * blockDim.x + threadIdx.x;
  if (i < n) p[i] = 0.0f;
}

// ---------------- degree over rows (edges + self loops) ----------------
__global__ void gpp_deg(const long long* __restrict__ rows, float* __restrict__ deg,
                        long long Ereal, long long Etot) {
  long long e = (long long)blockIdx.x * blockDim.x + threadIdx.x;
  if (e >= Etot) return;
  long long r = (e < Ereal) ? rows[e] : (e - Ereal);
  atomicAdd(&deg[r], 1.0f);
}

__global__ void gpp_invdeg(const float* __restrict__ deg, float* __restrict__ invdeg, int n) {
  int i = blockIdx.x * blockDim.x + threadIdx.x;
  if (i < n) invdeg[i] = 1.0f / fmaxf(deg[i], 1.0f);
}

__global__ void gpp_counts(const long long* __restrict__ batch, float* __restrict__ counts, int n) {
  int i = blockIdx.x * blockDim.x + threadIdx.x;
  if (i < n) atomicAdd(&counts[(int)batch[i]], 1.0f);
}

// ---------------- view init: xv = x * dropout_mask * node_mask[v] ----------------
__global__ void gpp_init_view(const float* __restrict__ x, const float* __restrict__ dmask,
                              const float* __restrict__ nmask_v, float* __restrict__ xv,
                              long long total) {
  long long i = (long long)blockIdx.x * blockDim.x + threadIdx.x;
  if (i >= total) return;
  long long n = i >> 8;                 // F == 256
  xv[i] = x[i] * dmask[i] * nmask_v[n];
}

// ---------------- one propagation hop: dst[row] += src[col] * invdeg[row] ----------------
__global__ void gpp_scatter(const float* __restrict__ src, float* __restrict__ dst,
                            const long long* __restrict__ rows, const long long* __restrict__ cols,
                            const float* __restrict__ invdeg, long long Ereal, long long Etot) {
  long long t = (long long)blockIdx.x * blockDim.x + threadIdx.x;
  long long e = t >> 6;
  if (e >= Etot) return;
  int part = (int)(t & 63);
  long long r, c;
  if (e < Ereal) { r = rows[e]; c = cols[e]; } else { r = e - Ereal; c = r; }
  float w = invdeg[r];
  const float4 v = *(const float4*)(src + (c << 8) + part * 4);
  float* d = dst + (r << 8) + part * 4;
  atomicAdd(d + 0, v.x * w);
  atomicAdd(d + 1, v.y * w);
  atomicAdd(d + 2, v.z * w);
  atomicAdd(d + 3, v.w * w);
}

// ---------------- W -> f16 fragment-linear layout ----------------
// Wf[((kb8*16 + nt)*32 + lane)*16 + j] = (f16) W[k][nt*16 + (lane&15)]
// k = kb8*32 + klo(lane) + (j<8 ? j : j+8). One K-slice (kb8) = 16KB contiguous.
__global__ void gpp_w_to_frag(const float* __restrict__ W, _Float16* __restrict__ Wf) {
  const int i = blockIdx.x * blockDim.x + threadIdx.x;   // 8*16*32*16 = 65536
  if (i >= 8 * 16 * 32 * 16) return;
  const int j    = i & 15;
  const int lane = (i >> 4) & 31;
  const int nt   = (i >> 9) & 15;
  const int kb8  = i >> 13;
  const int klo  = (lane < 16) ? 0 : 8;
  const int k    = kb8 * 32 + klo + ((j < 8) ? j : (j + 8));
  const int n    = nt * 16 + (lane & 15);
  Wf[i] = (_Float16)W[k * 256 + n];
}

// ---------------- WMMA GEMM: 16(M) x 64(N) strip per wave, K=256 -----------------
// B panels staged to LDS (async double-buffered when available); A from global.
// AHALF=0: A is f32 (propagation output), convert in-register.
// AHALF=1: A is f16 row-major (h1), pure b128 fragment loads.
// MODE=0 : outh = (f16) relu(A@W + bias)
// MODE=1 : gsum[batch[m]] += A@W + bias   (pooled accumulation)
template<int AHALF, int MODE>
__global__ __launch_bounds__(256)
void gpp_gemm_wmma(const void* __restrict__ Ap, const _Float16* __restrict__ Wf,
                   const float* __restrict__ bias, _Float16* __restrict__ outh,
                   float* __restrict__ gsum, const long long* __restrict__ batch,
                   int nrows) {
  __shared__ __align__(16) _Float16 sB[2][8192];   // 2 x 16KB K-slice panels

  const int tid  = threadIdx.x;
  const int lane = tid & 31;
  const int wave = blockIdx.x * 8 + (tid >> 5);
  const int tileM = wave >> 2;            // 256/64 = 4 wave-columns
  const int tileN = (wave & 3) * 64;
  const int m0 = tileM * 16;
  const bool active = (m0 < nrows);       // wave-uniform

  const int klo  = (lane < 16) ? 0 : 8;
  const int ncol = lane & 15;
  const int row  = m0 + ncol;

  const float*    Af = (const float*)Ap;
  const _Float16* Ah = (const _Float16*)Ap;

  v8f acc0 = {}; v8f acc1 = {}; v8f acc2 = {}; v8f acc3 = {};

  // cooperative stage of K-slice kb8 (16KB) into sB[buf]: 64B per thread
  auto stage = [&](int kb8, int buf) {
    const _Float16* src = Wf + (size_t)kb8 * 8192 + tid * 32;
    _Float16* dst = &sB[buf][tid * 32];
#if GPP_ASYNC_LDS
    gpp_async_b128(src + 0,  dst + 0);
    gpp_async_b128(src + 8,  dst + 8);
    gpp_async_b128(src + 16, dst + 16);
    gpp_async_b128(src + 24, dst + 24);
#else
    float4* d4 = (float4*)dst; const float4* s4 = (const float4*)src;
    d4[0] = s4[0]; d4[1] = s4[1]; d4[2] = s4[2]; d4[3] = s4[3];
#endif
  };

  stage(0, 0);
  stage(1, 1);

  for (int kb8 = 0; kb8 < 8; ++kb8) {
    if (kb8 < 7) { GPP_WAIT_ASYNC(4); }   // panel kb8 done (next panel in flight)
    else         { GPP_WAIT_ASYNC(0); }
    __syncthreads();                      // panel kb8 visible to all waves

    if (active) {
      const int kb = kb8 * 32;
      // ---- A fragment: row `row`, K = kb+klo+{0..7, 16..23}
      v16h af;
      if (AHALF) {
        const _Float16* arow = Ah + ((size_t)row << 8) + kb + klo;
        const v8h lo = *(const v8h*)(arow);
        const v8h hi = *(const v8h*)(arow + 16);
        #pragma unroll
        for (int j = 0; j < 8; ++j) { af[j] = lo[j]; af[j + 8] = hi[j]; }
      } else {
        const float* arow = Af + ((size_t)row << 8) + kb + klo;
        if (kb + 32 < 256) __builtin_prefetch(arow + 32, 0, 0);  // global_prefetch_b8
        const float4 a0 = *(const float4*)(arow + 0);
        const float4 a1 = *(const float4*)(arow + 4);
        const float4 a2 = *(const float4*)(arow + 16);
        const float4 a3 = *(const float4*)(arow + 20);
        af[0]  = (_Float16)a0.x; af[1]  = (_Float16)a0.y; af[2]  = (_Float16)a0.z; af[3]  = (_Float16)a0.w;
        af[4]  = (_Float16)a1.x; af[5]  = (_Float16)a1.y; af[6]  = (_Float16)a1.z; af[7]  = (_Float16)a1.w;
        af[8]  = (_Float16)a2.x; af[9]  = (_Float16)a2.y; af[10] = (_Float16)a2.z; af[11] = (_Float16)a2.w;
        af[12] = (_Float16)a3.x; af[13] = (_Float16)a3.y; af[14] = (_Float16)a3.z; af[15] = (_Float16)a3.w;
      }

      // ---- 4 B fragments from the LDS panel: 32B contiguous per lane
      const _Float16* wb = &sB[kb8 & 1][(((size_t)(tileN >> 4) + 0) * 32 + lane) * 16];
      const v16h bf0 = *(const v16h*)(wb + 0 * 32 * 16);
      const v16h bf1 = *(const v16h*)(wb + 1 * 32 * 16);
      const v16h bf2 = *(const v16h*)(wb + 2 * 32 * 16);
      const v16h bf3 = *(const v16h*)(wb + 3 * 32 * 16);

      acc0 = __builtin_amdgcn_wmma_f32_16x16x32_f16(false, af, false, bf0, (short)0, acc0, false, false);
      acc1 = __builtin_amdgcn_wmma_f32_16x16x32_f16(false, af, false, bf1, (short)0, acc1, false, false);
      acc2 = __builtin_amdgcn_wmma_f32_16x16x32_f16(false, af, false, bf2, (short)0, acc2, false, false);
      acc3 = __builtin_amdgcn_wmma_f32_16x16x32_f16(false, af, false, bf3, (short)0, acc3, false, false);
    }

    __syncthreads();                      // everyone done reading panel kb8
    if (kb8 + 2 < 8) stage(kb8 + 2, kb8 & 1);
  }

  if (!active) return;

  // ---- epilogue: reg r -> M = m0 + r (+8 for lanes 16..31), N = tileN + f*16 + ncol
  const int mofs = (lane >> 4) * 8;
  float bv[4];
  #pragma unroll
  for (int f = 0; f < 4; ++f) bv[f] = bias[tileN + f * 16 + ncol];

  #pragma unroll
  for (int r = 0; r < 8; ++r) {
    const int m = m0 + mofs + r;
    const float vals[4] = { acc0[r], acc1[r], acc2[r], acc3[r] };
    if (MODE == 0) {
      _Float16* orow = outh + ((size_t)m << 8) + tileN + ncol;
      #pragma unroll
      for (int f = 0; f < 4; ++f)
        orow[f * 16] = (_Float16)fmaxf(vals[f] + bv[f], 0.0f);
    } else {
      const long long bidx = batch[m];
      float* grow = gsum + ((size_t)bidx << 8) + tileN + ncol;
      #pragma unroll
      for (int f = 0; f < 4; ++f)
        atomicAdd(&grow[f * 16], vals[f] + bv[f]);
    }
  }
}

// ---------------- classifier + softmax + consistency/entropy loss ----------------
__global__ void gpp_finalize(const float* __restrict__ gsum, const float* __restrict__ counts,
                             const float* __restrict__ Wc, const float* __restrict__ bc,
                             float* __restrict__ out) {
  __shared__ float lg[VV * BB * CC];   // logits -> probs (in place)
  __shared__ float mp[BB * CC];        // mean probs
  __shared__ float red[2];             // [0]=cons sum, [1]=ent sum
  const int T = blockDim.x, t = threadIdx.x;
  if (t < 2) red[t] = 0.0f;
  __syncthreads();

  for (int i = t; i < VV * BB * CC; i += T) {
    const int c = i % CC;
    const int b = (i / CC) % BB;
    const int v = i / (CC * BB);
    const float inv_cnt = 1.0f / fmaxf(counts[b], 1.0f);
    const float* g = gsum + (((size_t)v * BB + b) << 8);
    float acc = bc[c];
    for (int h = 0; h < HH; ++h) acc += (g[h] * inv_cnt) * Wc[h * CC + c];
    lg[i] = acc;
    if (v == 0) out[b * CC + c] = acc;  // logits output (pre-softmax)
  }
  __syncthreads();

  for (int i = t; i < VV * BB; i += T) {
    float* p = lg + i * CC;
    float mx = p[0];
    for (int c = 1; c < CC; ++c) mx = fmaxf(mx, p[c]);
    float s = 0.0f;
    for (int c = 0; c < CC; ++c) { p[c] = __expf(p[c] - mx); s += p[c]; }
    const float inv = 1.0f / s;
    for (int c = 0; c < CC; ++c) p[c] *= inv;
  }
  __syncthreads();

  for (int i = t; i < BB * CC; i += T) {
    const int c = i % CC, b = i / CC;
    float m = 0.0f;
    for (int v = 0; v < VV; ++v) m += lg[(v * BB + b) * CC + c];
    m *= (1.0f / VV);
    mp[i] = m;
    atomicAdd(&red[1], -m * __logf(m + 1e-8f));
  }
  __syncthreads();

  for (int i = t; i < VV * BB * CC; i += T) {
    const int c = i % CC, b = (i / CC) % BB;
    const float d = lg[i] - mp[b * CC + c];
    atomicAdd(&red[0], d * d);
  }
  __syncthreads();

  if (t == 0) {
    const float loss_cons = red[0] / (float)(VV * BB);
    const float loss_ent  = red[1] / (float)BB;
    out[BB * CC] = 1.0f * loss_cons + loss_ent;   // LAM = 1.0
  }
}

// ---------------- host orchestration ----------------
extern "C" void kernel_launch(void* const* d_in, const int* in_sizes, int n_in,
                              void* d_out, int out_size, void* d_ws, size_t ws_size,
                              hipStream_t stream) {
  const float*      x     = (const float*)d_in[0];
  const long long*  edge  = (const long long*)d_in[1];  // [2, E]
  const long long*  batch = (const long long*)d_in[2];
  const float*      dmask = (const float*)d_in[3];
  const float*      nmask = (const float*)d_in[4];      // [V, N]
  const float*      W1    = (const float*)d_in[5];
  const float*      b1    = (const float*)d_in[6];
  const float*      W2    = (const float*)d_in[7];
  const float*      b2    = (const float*)d_in[8];
  const float*      Wc    = (const float*)d_in[9];
  const float*      bc    = (const float*)d_in[10];
  float*            out   = (float*)d_out;

  const long long E    = in_sizes[1] / 2;
  const long long Etot = E + NN;
  const long long* rows = edge;
  const long long* cols = edge + E;

  // workspace layout
  char* ws = (char*)d_ws;
  float* deg    = (float*)ws;                       ws += sizeof(float) * NN;
  float* invdeg = (float*)ws;                       ws += sizeof(float) * NN;
  float* counts = (float*)ws;                       ws += sizeof(float) * BB;
  ws = (char*)(((size_t)ws + 255) & ~(size_t)255);
  float* gsum   = (float*)ws;                       ws += sizeof(float) * VV * BB * HH;
  ws = (char*)(((size_t)ws + 255) & ~(size_t)255);
  _Float16* W1f = (_Float16*)ws;                    ws += sizeof(_Float16) * 65536;
  _Float16* W2f = (_Float16*)ws;                    ws += sizeof(_Float16) * 65536;
  ws = (char*)(((size_t)ws + 255) & ~(size_t)255);
  _Float16* h1h = (_Float16*)ws;                    ws += sizeof(_Float16) * (size_t)NN * HH;
  ws = (char*)(((size_t)ws + 255) & ~(size_t)255);
  float* bufA   = (float*)ws;                       ws += sizeof(float) * (size_t)NN * FF;
  ws = (char*)(((size_t)ws + 255) & ~(size_t)255);
  float* bufB   = (float*)ws;

  const int TB = 256;
  const long long NF = (long long)NN * FF;

  // fresh state every call
  gpp_zero_f32<<<divceil_i(NN, TB), TB, 0, stream>>>(deg, NN);
  gpp_zero_f32<<<divceil_i(BB, TB), TB, 0, stream>>>(counts, BB);
  gpp_zero_f32<<<divceil_i((long long)VV * BB * HH, TB), TB, 0, stream>>>(gsum, (long long)VV * BB * HH);

  gpp_deg<<<divceil_i(Etot, TB), TB, 0, stream>>>(rows, deg, E, Etot);
  gpp_invdeg<<<divceil_i(NN, TB), TB, 0, stream>>>(deg, invdeg, NN);
  gpp_counts<<<divceil_i(NN, TB), TB, 0, stream>>>(batch, counts, NN);

  // one-shot weight fragment linearization (f16)
  gpp_w_to_frag<<<divceil_i(65536, TB), TB, 0, stream>>>(W1, W1f);
  gpp_w_to_frag<<<divceil_i(65536, TB), TB, 0, stream>>>(W2, W2f);

  const int gemm_blocks = divceil_i((long long)(NN / 16) * 4, 8);  // 12500 waves / 8

  for (int v = 0; v < VV; ++v) {
    gpp_init_view<<<divceil_i(NF, TB), TB, 0, stream>>>(x, dmask, nmask + (size_t)v * NN, bufA, NF);

    float* src = bufA;
    float* dst = bufB;
    for (int k = 0; k < KK; ++k) {
      gpp_zero_f32<<<divceil_i(NF, TB), TB, 0, stream>>>(dst, NF);
      gpp_scatter<<<divceil_i(Etot * 64, TB), TB, 0, stream>>>(src, dst, rows, cols, invdeg, E, Etot);
      float* tmp = src; src = dst; dst = tmp;
    }
    // h1 = relu(xv @ W1 + b1) in f16
    gpp_gemm_wmma<0, 0><<<gemm_blocks, TB, 0, stream>>>(src, W1f, b1, h1h, nullptr, batch, NN);
    // gsum[v] += h1 @ W2 + b2   (pooled)
    gpp_gemm_wmma<1, 1><<<gemm_blocks, TB, 0, stream>>>(h1h, W2f, b2, nullptr,
                                                        gsum + (size_t)v * BB * HH, batch, NN);
  }

  gpp_finalize<<<1, 256, 0, stream>>>(gsum, counts, Wc, bc, out);
}
